// GraphAttentionConvolution_28784870818074
// MI455X (gfx1250) — matrix operand order repaired and checked by reading, not compiled, not run on hardware
//
#include <hip/hip_runtime.h>
#include <stdint.h>

// ---------------------------------------------------------------------------
// GraphAttentionConvolution for MI455X (gfx1250, wave32, WMMA, TDM)
//   stage1: x1  = bf16( (att * adj) @ inputs )          8192x8192x256 GEMM
//   stage2: out = relu( x1 @ W + b )                    8192x256x256  GEMM
// Roofline: 512 MB of att+adj streamed once @ 23.3 TB/s ~= 22 us floor;
// bf16 WMMA (16x16x32) pushes the 34 GFLOP of matrix math far below that.
// LDS row stride is 20 DWORDs (16 data + 4 pad): 16B-aligned for ds_load_b128,
// bank-conflict-free (20*l mod 64 distinct for l=0..15), and producible by the
// TDM pad engine (pad_interval=16 DW, pad_amount=4 DW).
// ---------------------------------------------------------------------------

#define NN 8192   // nodes (rows and K of stage 1)
#define F  256    // features (in == out)
#define LSTR 20   // LDS row stride in dwords (16 data + 4 pad)

typedef __attribute__((ext_vector_type(16))) __bf16 v16bf;
typedef __attribute__((ext_vector_type(8)))  float  v8f;
typedef __attribute__((ext_vector_type(4)))  unsigned int v4u;
typedef __attribute__((ext_vector_type(8)))  int    v8i;
typedef __attribute__((ext_vector_type(4)))  int    v4i;

union Frag { uint4 q[2]; unsigned u[8]; v16bf v; };

__device__ __forceinline__ unsigned short f32_to_bf16(float f) {
  unsigned u = __builtin_bit_cast(unsigned, f);
  u += 0x7FFFu + ((u >> 16) & 1u);          // round-to-nearest-even
  return (unsigned short)(u >> 16);
}
__device__ __forceinline__ unsigned pack2_bf16(float lo, float hi) {
  return (unsigned)f32_to_bf16(lo) | ((unsigned)f32_to_bf16(hi) << 16);
}

__device__ __forceinline__ unsigned lds_offset(const void* p) {
  // low 32 bits of a flat shared address == LDS byte offset (aperture in high bits)
  return (unsigned)(unsigned long long)reinterpret_cast<uintptr_t>(p);
}

// 2D tile DMA via Tensor Data Mover. Descriptor per CDNA5 ISA.
// stride0/dims in elements of (1<<ds_log) bytes.
// pad_enable: insert pad_amount+1 DWORDs after every 2<<pad_interval DWORDs.
__device__ __forceinline__ void tdm_load_2d(unsigned lds_addr, const void* gptr,
                                            unsigned ds_log,
                                            unsigned tensor_d0, unsigned tensor_d1,
                                            unsigned stride0,
                                            unsigned tile_d0, unsigned tile_d1,
                                            unsigned pad_enable, unsigned pad_interval,
                                            unsigned pad_amount) {
  unsigned long long ga = (unsigned long long)reinterpret_cast<uintptr_t>(gptr);
  v4u g0;
  g0.x = 1u;                                       // count=1, user descriptor
  g0.y = lds_addr;                                 // LDS byte address
  g0.z = (unsigned)ga;                             // global_addr[31:0]
  g0.w = (unsigned)(ga >> 32) | (2u << 30);        // global_addr[56:32] | type=2
  v8i g1;
  g1[0] = (int)((ds_log << 16) | (pad_enable << 20) | (pad_interval << 22) |
                (pad_amount << 25));
  g1[1] = (int)((tensor_d0 & 0xFFFFu) << 16);                      // dim0 lo16
  g1[2] = (int)((tensor_d0 >> 16) | ((tensor_d1 & 0xFFFFu) << 16));// dim0 hi | dim1 lo
  g1[3] = (int)((tensor_d1 >> 16) | (tile_d0 << 16));              // dim1 hi | tile0
  g1[4] = (int)(tile_d1 & 0xFFFFu);                                // tile1 | tile2=0
  g1[5] = (int)stride0;                                            // dim0_stride lo32
  g1[6] = 0;                                                       // stride hi | dim1_stride
  g1[7] = 0;
  v4i z4 = {0, 0, 0, 0};                           // groups 2/3 unused (2D tile)
  v8i z8 = {0, 0, 0, 0, 0, 0, 0, 0};               // trailing group (6-arg variant)
  __builtin_amdgcn_tensor_load_to_lds(g0, g1, z4, z4, z8, 0);
}

// ---------------------------------------------------------------------------
// prep: pack f32 [2*Ptot][Ntot] into bf16-K-pair dwords, TRANSPOSED to
// dst[Ntot][Ptot] so GEMM B-fragments are contiguous. Tiled via LDS so both
// the reads and the writes are fully coalesced. 64x64 dword tiles, 256 thr.
// ---------------------------------------------------------------------------
__global__ __launch_bounds__(256) void gat_pack_t(const float* __restrict__ src,
                                                  unsigned* __restrict__ dst,
                                                  int Ptot, int Ntot, int ptiles) {
  __shared__ unsigned s[64][65];
  const int p0 = (blockIdx.x % ptiles) * 64;
  const int n0 = (blockIdx.x / ptiles) * 64;
  const int t = threadIdx.x;
  {
    const int c = t & 63, pl0 = t >> 6;
    for (int i = 0; i < 16; ++i) {
      const int pl = pl0 + i * 4;
      const float lo = src[(size_t)(2 * (p0 + pl)) * Ntot + n0 + c];
      const float hi = src[(size_t)(2 * (p0 + pl) + 1) * Ntot + n0 + c];
      s[c][pl] = pack2_bf16(lo, hi);
    }
  }
  __syncthreads();
  {
    const int pl = t & 63, cr0 = t >> 6;
    for (int i = 0; i < 16; ++i) {
      const int cr = cr0 + i * 4;
      dst[(size_t)(n0 + cr) * Ptot + p0 + pl] = s[cr][pl];
    }
  }
}

// ---------------------------------------------------------------------------
// stage1: 32 rows x 256 cols per block (4 waves, each 32x64), K stepped by 32.
// Double-buffered software pipeline, ONE barrier per k-step:
//   iter i: [wait tensorcnt][barrier][fill A_{i+1} + issue TDM B_{i+1}][compute i]
// A tile: att*adj -> bf16 pairs via VALU.  B tile: TDM from x_pkT[N][Kpair].
// ---------------------------------------------------------------------------
__global__ __launch_bounds__(128) void gat_stage1(const float* __restrict__ att,
                                                  const int* __restrict__ adj,
                                                  const unsigned* __restrict__ x_pkT,
                                                  unsigned short* __restrict__ x1) {
  __shared__ __align__(16) unsigned a_lds[2][32 * LSTR];   // [m][20 dw]
  __shared__ __align__(16) unsigned b_lds[2][256 * LSTR];  // [n][20 dw]

  const int t    = threadIdx.x;
  const int wv   = t >> 5;
  const int lane = t & 31;
  const int hi16 = lane >> 4;
  const int l16  = lane & 15;
  const int m0   = blockIdx.x * 32;

  v8f acc[2][4] = {};

  const int am = t >> 2;                 // 0..31 : row within tile
  const int ak = (t & 3) * 8;            // 0,8,16,24 : k chunk within tile
  const float* arow = att + (size_t)(m0 + am) * NN + ak;
  const int*   jrow = adj + (size_t)(m0 + am) * NN + ak;

  auto fillA = [&](int buf, int kk) {
    const float4 a0 = ((const float4*)(arow + kk))[0];
    const float4 a1 = ((const float4*)(arow + kk))[1];
    const int4   j0 = ((const int4*)(jrow + kk))[0];
    const int4   j1 = ((const int4*)(jrow + kk))[1];
    uint4 v;
    v.x = pack2_bf16(a0.x * (float)j0.x, a0.y * (float)j0.y);
    v.y = pack2_bf16(a0.z * (float)j0.z, a0.w * (float)j0.w);
    v.z = pack2_bf16(a1.x * (float)j1.x, a1.y * (float)j1.y);
    v.w = pack2_bf16(a1.z * (float)j1.z, a1.w * (float)j1.w);
    *(uint4*)&a_lds[buf][am * LSTR + (ak >> 1)] = v;   // aligned ds_store_b128
    __builtin_prefetch(arow + kk + 32, 0, 0);
    __builtin_prefetch(jrow + kk + 32, 0, 0);
  };
  auto tdmB = [&](int buf, int kk) {     // 256 rows x 16 dw of packed inputs
    tdm_load_2d(lds_offset(&b_lds[buf][0]), x_pkT + (kk >> 1),
                /*ds=4B*/2, /*d0*/NN / 2, /*d1*/F, /*stride*/NN / 2,
                /*tile0*/16, /*tile1*/F, /*pad*/1, /*16dw*/3, /*+4dw*/3);
  };

  // prologue: stage tile 0
  if (wv == 0) tdmB(0, 0);
  fillA(0, 0);

  for (int it = 0; it < NN / 32; ++it) {
    const int cur = it & 1, nxt = cur ^ 1;
    if (wv == 0) __builtin_amdgcn_s_wait_tensorcnt(0);
    __syncthreads();
    if (it + 1 < NN / 32) {              // stage tile i+1 while computing tile i
      fillA(nxt, (it + 1) * 32);
      if (wv == 0) tdmB(nxt, (it + 1) * 32);
    }

    Frag a[2], b[4];
    for (int r = 0; r < 2; ++r) {        // A: two aligned b128 per fragment
      const uint4* ap = (const uint4*)&a_lds[cur][(r * 16 + l16) * LSTR + hi16 * 4];
      a[r].q[0] = ap[0];                 // K/2 = hi16*4 + 0..3
      a[r].q[1] = ap[2];                 // K/2 = 8 + hi16*4 + 0..3
    }
    for (int c = 0; c < 4; ++c) {        // B: two aligned b128 per fragment
      const uint4* bp =
          (const uint4*)&b_lds[cur][(wv * 64 + c * 16 + l16) * LSTR + hi16 * 4];
      b[c].q[0] = bp[0];
      b[c].q[1] = bp[2];
    }
    for (int r = 0; r < 2; ++r)
      for (int c = 0; c < 4; ++c)
        acc[r][c] = __builtin_amdgcn_wmma_f32_16x16x32_bf16(
            false, a[r].v, false, b[c].v, (short)0, acc[r][c], false, false);
  }

  for (int r = 0; r < 2; ++r)
    for (int c = 0; c < 4; ++c)
      for (int e = 0; e < 8; ++e) {
        const int M = m0 + r * 16 + hi16 * 8 + e;
        const int N = wv * 64 + c * 16 + l16;
        x1[(size_t)M * F + N] = f32_to_bf16(acc[r][c][e]);
      }
}

// ---------------------------------------------------------------------------
// stage2: out = relu(x1 @ W + b). Both tiles via TDM with the pad engine
// producing the 20-dword LDS stride. Only 8 k-steps -> simple schedule.
// ---------------------------------------------------------------------------
__global__ __launch_bounds__(128) void gat_stage2(const unsigned short* __restrict__ x1,
                                                  const unsigned* __restrict__ w_pkT,
                                                  const float* __restrict__ bias,
                                                  float* __restrict__ out) {
  __shared__ __align__(16) unsigned a_lds[32 * LSTR];
  __shared__ __align__(16) unsigned b_lds[256 * LSTR];

  const int t    = threadIdx.x;
  const int wv   = t >> 5;
  const int lane = t & 31;
  const int hi16 = lane >> 4;
  const int l16  = lane & 15;
  const int m0   = blockIdx.x * 32;

  v8f acc[2][4] = {};

  for (int kk = 0; kk < F; kk += 32) {
    __syncthreads();
    if (wv == 0) {
      // A: 32x32 bf16 tile of x1 (row stride 256), LDS rows padded 16dw+4dw
      tdm_load_2d(lds_offset(a_lds), x1 + (size_t)m0 * F + kk,
                  /*ds=2B*/1, /*d0*/F, /*d1*/NN, /*stride*/F,
                  /*tile0*/32, /*tile1*/32, /*pad*/1, 3, 3);
      // B: 256 rows x 16 dw of packed W (w_pkT[N][Kpair])
      tdm_load_2d(lds_offset(b_lds), w_pkT + (kk >> 1),
                  /*ds=4B*/2, /*d0*/F / 2, /*d1*/F, /*stride*/F / 2,
                  /*tile0*/16, /*tile1*/F, /*pad*/1, 3, 3);
      __builtin_amdgcn_s_wait_tensorcnt(0);
    }
    __syncthreads();

    Frag a[2], b[4];
    for (int r = 0; r < 2; ++r) {
      const uint4* ap = (const uint4*)&a_lds[(r * 16 + l16) * LSTR + hi16 * 4];
      a[r].q[0] = ap[0];
      a[r].q[1] = ap[2];
    }
    for (int c = 0; c < 4; ++c) {
      const uint4* bp = (const uint4*)&b_lds[(wv * 64 + c * 16 + l16) * LSTR + hi16 * 4];
      b[c].q[0] = bp[0];
      b[c].q[1] = bp[2];
    }
    for (int r = 0; r < 2; ++r)
      for (int c = 0; c < 4; ++c)
        acc[r][c] = __builtin_amdgcn_wmma_f32_16x16x32_bf16(
            false, a[r].v, false, b[c].v, (short)0, acc[r][c], false, false);
  }

  for (int c = 0; c < 4; ++c) {
    const int N = wv * 64 + c * 16 + l16;
    const float bN = bias[N];
    for (int r = 0; r < 2; ++r)
      for (int e = 0; e < 8; ++e) {
        const int M = m0 + r * 16 + hi16 * 8 + e;
        out[(size_t)M * F + N] = fmaxf(acc[r][c][e] + bN, 0.0f);
      }
  }
}

// ---------------------------------------------------------------------------
extern "C" void kernel_launch(void* const* d_in, const int* in_sizes, int n_in,
                              void* d_out, int out_size, void* d_ws, size_t ws_size,
                              hipStream_t stream) {
  (void)in_sizes; (void)n_in; (void)out_size; (void)ws_size;
  const int*   adj    = (const int*)d_in[0];     // adjacency [8192][8192] i32
  const float* inputs = (const float*)d_in[1];   // [8192][256] f32
  const float* att    = (const float*)d_in[2];   // [8192][8192] f32
  const float* W      = (const float*)d_in[3];   // [256][256] f32
  const float* bias   = (const float*)d_in[4];   // [256] f32
  float* out = (float*)d_out;

  char* ws = (char*)d_ws;
  unsigned*       x_pkT = (unsigned*)ws;                                  // [256][4096] 4 MB
  unsigned*       w_pkT = (unsigned*)(ws + (size_t)4 * 1024 * 1024);      // [256][128] 128 KB
  unsigned short* x1    = (unsigned short*)(ws + (size_t)4 * 1024 * 1024 + 128 * 1024); // 4 MB

  gat_pack_t<<<256, 256, 0, stream>>>(inputs, x_pkT, NN / 2, F, (NN / 2) / 64);
  gat_pack_t<<<8,   256, 0, stream>>>(W,      w_pkT, F / 2,  F, (F / 2) / 64);
  gat_stage1<<<NN / 32, 128, 0, stream>>>(att, adj, x_pkT, x1);
  gat_stage2<<<NN / 32, 128, 0, stream>>>(x1, w_pkT, bias, out);
}
